// MultiHeadAttn_75282186764964
// MI455X (gfx1250) — matrix operand order repaired and checked by reading, not compiled, not run on hardware
//
#include <hip/hip_runtime.h>
#include <hip/hip_bf16.h>

// ---------------------------------------------------------------------------
// MHA for MI455X (gfx1250): f16 WMMA everywhere, fp32 accumulate.
//   1) Q/K/V projections: fp32 GEMM, double-buffered LDS, f16 WMMA tiles
//   2) flash attention:   online softmax, 64-key blocks, async-to-LDS staging
//   3) output projection: f16 ctx x fp32 Wo -> fp32 d_out
// ---------------------------------------------------------------------------

typedef __attribute__((ext_vector_type(16))) _Float16 v16h;
typedef __attribute__((ext_vector_type(8)))  _Float16 v8h;
typedef __attribute__((ext_vector_type(4)))  _Float16 v4h;
typedef __attribute__((ext_vector_type(8)))  float    v8f;
typedef __attribute__((ext_vector_type(4)))  int      v4i;

#define DIM   1024
#define NHEAD 16
#define HD    64
#define SEQ   4096

// ---- gfx1250 async global->LDS copy (guarded; fallback = plain load) ------
#if defined(__has_builtin)
#  if __has_builtin(__builtin_amdgcn_global_load_async_to_lds_b128) && \
      __has_builtin(__builtin_amdgcn_s_wait_asynccnt)
#    define HAVE_ASYNC_LDS 1
#  endif
#endif
#ifndef HAVE_ASYNC_LDS
#  define HAVE_ASYNC_LDS 0
#endif

#if HAVE_ASYNC_LDS
#define GAS __attribute__((address_space(1)))
#define LAS __attribute__((address_space(3)))
__device__ __forceinline__ void async_cp16(const _Float16* g, _Float16* l) {
  // Builtin signature (clang diag): (int4 AS1*, int4 AS3*, imm offset, imm cpol)
  GAS v4i* gp = (GAS v4i*)(GAS void*)(void*)const_cast<_Float16*>(g);
  LAS v4i* lp = (LAS v4i*)(LAS void*)(void*)l;
  __builtin_amdgcn_global_load_async_to_lds_b128(gp, lp, 0, 0);
}
#endif

// ---- WMMA helper ----------------------------------------------------------
__device__ __forceinline__ v8f wmma32(v16h a, v16h b, v8f c) {
  // D = A(16x32 f16) * B(32x16 f16) + C(16x16 f32)
  return __builtin_amdgcn_wmma_f32_16x16x32_f16(
      false, a, false, b, (short)0, c, false, false);
}

// ---- fragment loaders from row-major 16-bit LDS tiles ---------------------
// A fragment (ISA 7.12.2): lane<16 -> row=lane, K {0..7,16..23};
//                          lane>=16 -> row=lane-16, K {8..15,24..31}
__device__ __forceinline__ v16h frag_a(const _Float16* tile, int ld, int lane) {
  int r  = lane & 15;
  int kb = (lane >> 4) * 8;
  const _Float16* p = tile + r * ld + kb;
  v8h lo = *(const v8h*)(p);
  v8h hi = *(const v8h*)(p + 16);
  return __builtin_shufflevector(lo, hi, 0,1,2,3,4,5,6,7,8,9,10,11,12,13,14,15);
}
// B fragment (from B^T stored row-major [n][k]): lane<16 -> col=lane, K {0..7,16..23};
// lane>=16 -> col=lane-16, K {8..15,24..31}.
__device__ __forceinline__ v16h frag_b(const _Float16* tile, int ld, int lane) {
  int n  = lane & 15;
  int kb = (lane >> 4) * 8;
  const _Float16* p = tile + n * ld + kb;
  v8h lo = *(const v8h*)(p);
  v8h hi = *(const v8h*)(p + 16);
  return __builtin_shufflevector(lo, hi, 0,1,2,3,4,5,6,7,8,9,10,11,12,13,14,15);
}

// ---- typed 4-element loaders / converters ---------------------------------
template <typename T> struct Vec4;
template <> struct Vec4<float>    { using type = float4; };
template <> struct Vec4<_Float16> { using type = v4h;    };

__device__ __forceinline__ float4 ld4(const float* p)    { return *(const float4*)p; }
__device__ __forceinline__ v4h    ld4(const _Float16* p) { return *(const v4h*)p;    }
__device__ __forceinline__ v4h tohalf4(float4 v) {
  v4h r;
  r[0] = (_Float16)v.x; r[1] = (_Float16)v.y;
  r[2] = (_Float16)v.z; r[3] = (_Float16)v.w;
  return r;
}
__device__ __forceinline__ v4h tohalf4(v4h v) { return v; }

// ---------------------------------------------------------------------------
// GEMM: Out[M,N] = A[M,K] @ W[N,K]^T + bias[N]
// Block tile 128x128, BK=32, 256 threads = 8 waves, wave tile 64x32.
// Double-buffered LDS; global fetch of tile k+1 overlaps WMMA on tile k.
// ---------------------------------------------------------------------------
#define BM  128
#define BN  128
#define BK  32
#define LDT 40   // halves per LDS row (80 B, 16B-aligned stride)

template <typename AT, typename OT>
__global__ __launch_bounds__(256) void gemm_bias_wmma(
    const AT* __restrict__ A, const float* __restrict__ W,
    const float* __restrict__ bias, OT* __restrict__ Out,
    int M, int N, int K)
{
  __shared__ __align__(16) _Float16 sA[2][BM][LDT];
  __shared__ __align__(16) _Float16 sB[2][BN][LDT];   // = B^T tile (W rows)

  const int t    = threadIdx.x;
  const int lane = t & 31;
  const int wid  = t >> 5;
  const int wm   = wid >> 2;                 // 0..1 : 64-row slab
  const int wn   = wid & 3;                  // 0..3 : 32-col slab
  const int bm0  = blockIdx.x * BM;
  const int bn0  = blockIdx.y * BN;

  int ldrow[4], ldcol[4];
  #pragma unroll
  for (int i = 0; i < 4; i++) {
    int idx  = t + i * 256;                  // 0..1023 quads
    ldrow[i] = idx >> 3;                     // 0..127
    ldcol[i] = (idx & 7) * 4;                // 0..28
  }

  const v8f vzero = {0.f,0.f,0.f,0.f,0.f,0.f,0.f,0.f};
  v8f acc[4][2];
  #pragma unroll
  for (int i = 0; i < 4; i++)
    #pragma unroll
    for (int j = 0; j < 2; j++) acc[i][j] = vzero;

  typename Vec4<AT>::type rA[4];
  float4                  rB[4];

  auto fetch = [&](int k0) {
    #pragma unroll
    for (int i = 0; i < 4; i++) {
      rA[i] = ld4(&A[(size_t)(bm0 + ldrow[i]) * K + k0 + ldcol[i]]);
      rB[i] = ld4(&W[(size_t)(bn0 + ldrow[i]) * K + k0 + ldcol[i]]);
    }
  };
  auto stage = [&](int buf) {
    #pragma unroll
    for (int i = 0; i < 4; i++) {
      *(v4h*)&sA[buf][ldrow[i]][ldcol[i]] = tohalf4(rA[i]);
      *(v4h*)&sB[buf][ldrow[i]][ldcol[i]] = tohalf4(rB[i]);
    }
  };

  fetch(0);
  int buf = 0;
  for (int k0 = 0; k0 < K; k0 += BK) {
    stage(buf);
    __syncthreads();
    if (k0 + BK < K) fetch(k0 + BK);        // overlaps with WMMA below

    v16h af[4], bf[2];
    #pragma unroll
    for (int mt = 0; mt < 4; mt++) af[mt] = frag_a(&sA[buf][wm*64 + mt*16][0], LDT, lane);
    #pragma unroll
    for (int nt = 0; nt < 2; nt++) bf[nt] = frag_b(&sB[buf][wn*32 + nt*16][0], LDT, lane);
    #pragma unroll
    for (int mt = 0; mt < 4; mt++)
      #pragma unroll
      for (int nt = 0; nt < 2; nt++)
        acc[mt][nt] = wmma32(af[mt], bf[nt], acc[mt][nt]);
    buf ^= 1;
  }

  // Epilogue: C layout (ISA 7.12.2): col = lane&15, rows = (lane>>4)*8 + vgpr
  const int cn    = lane & 15;
  const int rbase = (lane >> 4) * 8;
  #pragma unroll
  for (int mt = 0; mt < 4; mt++) {
    #pragma unroll
    for (int nt = 0; nt < 2; nt++) {
      int   col = bn0 + wn*32 + nt*16 + cn;
      float b   = bias[col];
      #pragma unroll
      for (int r = 0; r < 8; r++) {
        int row = bm0 + wm*64 + mt*16 + rbase + r;
        Out[(size_t)row * N + col] = (OT)(acc[mt][nt][r] + b);
      }
    }
  }
}

// ---------------------------------------------------------------------------
// Flash attention: 4 waves/block, wave owns 16 query rows x full head (64).
// Key blocks of 64 staged in LDS (K via async-to-LDS when available);
// online softmax; ctx accumulated in v8f registers.
// ---------------------------------------------------------------------------
__global__ __launch_bounds__(128) void flash_attn_wmma(
    const _Float16* __restrict__ Qp, const _Float16* __restrict__ Kp,
    const _Float16* __restrict__ Vp, _Float16* __restrict__ Ctx)
{
  __shared__ __align__(16) _Float16 sK [64][72];      // [key][d]   == B^T for QK^T
  __shared__ __align__(16) _Float16 sVt[64][72];      // [d][key]   == B^T for P*V
  __shared__ __align__(16) _Float16 sP [4][16][72];   // per-wave probs (A matrix)
  __shared__ float sS [4][16][68];                    // per-wave raw scores
  __shared__ float sBc[4][16];                        // per-row broadcast (alpha / 1/l)

  const int t     = threadIdx.x;
  const int lane  = t & 31;
  const int wid   = t >> 5;
  const int h     = blockIdx.y;
  const int q0    = blockIdx.x * 64;
  const int hc0   = h * HD;
  const int cn    = lane & 15;
  const int rbase = (lane >> 4) * 8;

  // Q tile (16x64) for this wave as two resident A-fragments
  const int qrow = q0 + wid * 16 + (lane & 15);
  const int kbA  = (lane >> 4) * 8;
  v16h qf[2];
  #pragma unroll
  for (int c = 0; c < 2; c++) {
    const _Float16* p = Qp + (size_t)qrow * DIM + hc0 + c*32 + kbA;
    v8h lo = *(const v8h*)(p);
    v8h hi = *(const v8h*)(p + 16);
    qf[c] = __builtin_shufflevector(lo, hi, 0,1,2,3,4,5,6,7,8,9,10,11,12,13,14,15);
  }

  const v8f vzero = {0.f,0.f,0.f,0.f,0.f,0.f,0.f,0.f};
  v8f o[4];
  #pragma unroll
  for (int nt = 0; nt < 4; nt++) o[nt] = vzero;

  float m_run = -1e30f, l_run = 0.f;
  const int srow  = lane & 15;            // softmax row for this lane
  const int shalf = (lane >> 4) * 32;     // column half (2 lanes per row)

  for (int kb0 = 0; kb0 < SEQ; kb0 += 64) {
    __syncthreads();
    // Cooperative staging: K rows straight (async path), V transposed.
    #pragma unroll
    for (int i = 0; i < 4; i++) {
      int idx = t + i * 128;              // 0..511
      int r   = idx >> 3;                 // 0..63
      int c8  = (idx & 7) * 8;            // 0..56
      const _Float16* kg = Kp + (size_t)(kb0 + r) * DIM + hc0 + c8;
#if HAVE_ASYNC_LDS
      async_cp16(kg, &sK[r][c8]);         // global_load_async_to_lds_b128
#else
      *(v8h*)&sK[r][c8] = *(const v8h*)kg;
#endif
      v8h vv = *(const v8h*)(Vp + (size_t)(kb0 + r) * DIM + hc0 + c8);
      #pragma unroll
      for (int e = 0; e < 8; e++) sVt[c8 + e][r] = vv[e];
    }
#if HAVE_ASYNC_LDS
    __builtin_amdgcn_s_wait_asynccnt(0);
#endif
    __syncthreads();

    // S = (Q K^T) / sqrt(64) : 4 column tiles x 2 K-chunks = 8 WMMA
    #pragma unroll
    for (int nt = 0; nt < 4; nt++) {
      v8f s = vzero;
      #pragma unroll
      for (int kc = 0; kc < 2; kc++)
        s = wmma32(qf[kc], frag_b(&sK[nt*16][kc*32], 72, lane), s);
      #pragma unroll
      for (int r = 0; r < 8; r++)
        sS[wid][rbase + r][nt*16 + cn] = s[r] * 0.125f;
    }
    __syncthreads();

    // Online softmax: lanes L and L+16 split the 64 columns of row L&15.
    const float* srp = &sS[wid][srow][shalf];
    float lmax = -1e30f;
    #pragma unroll
    for (int c = 0; c < 32; c++) lmax = fmaxf(lmax, srp[c]);
    float omax  = fmaxf(lmax, __shfl_xor(lmax, 16, 32));
    float mnew  = fmaxf(m_run, omax);
    float alpha = __expf(m_run - mnew);
    float lsum  = 0.f;
    #pragma unroll
    for (int c = 0; c < 32; c++) {
      float p = __expf(srp[c] - mnew);
      sP[wid][srow][shalf + c] = (_Float16)p;
      lsum += p;
    }
    lsum += __shfl_xor(lsum, 16, 32);
    l_run = l_run * alpha + lsum;
    m_run = mnew;
    if (lane < 16) sBc[wid][srow] = alpha;
    __syncthreads();

    // Rescale running context by alpha, then O += P * V (8 WMMA)
    float al[8];
    #pragma unroll
    for (int r = 0; r < 8; r++) al[r] = sBc[wid][rbase + r];
    #pragma unroll
    for (int nt = 0; nt < 4; nt++)
      #pragma unroll
      for (int r = 0; r < 8; r++) o[nt][r] *= al[r];

    #pragma unroll
    for (int kc = 0; kc < 2; kc++) {
      v16h pf = frag_a(&sP[wid][0][kc*32], 72, lane);
      #pragma unroll
      for (int nt = 0; nt < 4; nt++)
        o[nt] = wmma32(pf, frag_b(&sVt[nt*16][kc*32], 72, lane), o[nt]);
    }
  }

  // Normalize by running sum and write f16 context.
  __syncthreads();
  if (lane < 16) sBc[wid][srow] = 1.f / l_run;
  __syncthreads();
  float inv[8];
  #pragma unroll
  for (int r = 0; r < 8; r++) inv[r] = sBc[wid][rbase + r];
  #pragma unroll
  for (int nt = 0; nt < 4; nt++)
    #pragma unroll
    for (int r = 0; r < 8; r++) {
      int row = q0 + wid*16 + rbase + r;
      Ctx[(size_t)row * DIM + hc0 + nt*16 + cn] = (_Float16)(o[nt][r] * inv[r]);
    }
}

// ---------------------------------------------------------------------------
extern "C" void kernel_launch(void* const* d_in, const int* in_sizes, int n_in,
                              void* d_out, int out_size, void* d_ws, size_t ws_size,
                              hipStream_t stream) {
  (void)in_sizes; (void)n_in; (void)out_size; (void)ws_size;
  const float* query = (const float*)d_in[0];
  const float* key   = (const float*)d_in[1];
  const float* value = (const float*)d_in[2];
  const float* Wq    = (const float*)d_in[3];
  const float* bq    = (const float*)d_in[4];
  const float* Wk    = (const float*)d_in[5];
  const float* bk    = (const float*)d_in[6];
  const float* Wv    = (const float*)d_in[7];
  const float* bv    = (const float*)d_in[8];
  const float* Wo    = (const float*)d_in[9];
  const float* bo    = (const float*)d_in[10];

  // Workspace: Q/K/V projections + attention context, all f16 (4 x 8 MB).
  _Float16* Qp = (_Float16*)d_ws;
  _Float16* Kp = Qp + (size_t)SEQ * DIM;
  _Float16* Vp = Kp + (size_t)SEQ * DIM;
  _Float16* Ct = Vp + (size_t)SEQ * DIM;

  dim3 gg(SEQ / BM, DIM / BN);
  gemm_bias_wmma<float, _Float16><<<gg, 256, 0, stream>>>(query, Wq, bq, Qp, SEQ, DIM, DIM);
  gemm_bias_wmma<float, _Float16><<<gg, 256, 0, stream>>>(key,   Wk, bk, Kp, SEQ, DIM, DIM);
  gemm_bias_wmma<float, _Float16><<<gg, 256, 0, stream>>>(value, Wv, bv, Vp, SEQ, DIM, DIM);

  flash_attn_wmma<<<dim3(SEQ / 64, NHEAD), 128, 0, stream>>>(Qp, Kp, Vp, Ct);

  gemm_bias_wmma<_Float16, float><<<gg, 256, 0, stream>>>(Ct, Wo, bo, (float*)d_out, SEQ, DIM, DIM);
}
